// Jin_2310692405390
// MI455X (gfx1250) — matrix operand order repaired and checked by reading, not compiled, not run on hardware
//
#include <hip/hip_runtime.h>
#include <hip/hip_bf16.h>
#include <math.h>

// ---------------------------------------------------------------------------
// GraphMAE-style forward for MI455X (gfx1250, wave32, WMMA).
// All GEMMs run through v_wmma_f32_16x16x32_f16 (f32->f16 tile conversion in
// LDS, f32 accumulate). B is staged TRANSPOSED in LDS so each lane's WMMA
// fragment is contiguous (2 x ds_load_b128). k_gemm is specialized at
// compile time on (AMODE, BTRANS, REDMODE) so the staging loops are
// branch-free; interior M-tiles take an unguarded fast path.
// Discriminator score matrices (6000x6000) are never materialized:
// -log(sigmoid(.)+eps) is reduced inside the GEMM epilogue.
// ---------------------------------------------------------------------------

typedef __attribute__((ext_vector_type(16))) _Float16 v16h;
typedef __attribute__((ext_vector_type(8)))  _Float16 v8h;
typedef __attribute__((ext_vector_type(8)))  float    v8f;

#define TBM 128
#define TBN 64
#define TBK 32
#define LDA_S 40   // halves per As row (80B, 16B-aligned rows)
#define LDBT  40   // halves per BsT row (80B, 16B-aligned rows)

// A-loader modes
#define AM_PLAIN  0
#define AM_GATHER 1
#define AM_POSTOK 2
#define AM_NEGSHF 3
// reduction modes
#define RED_NONE 0
#define RED_POS  1   // acc += -log(sigmoid(c)+eps)
#define RED_NEG  2   // acc += -log(1-sigmoid(c)+eps)

union PK4 { _Float16 h[4]; uint2 u; };

__device__ __forceinline__ void pk_store(_Float16* dstp, float4 v) {
    PK4 pk;
    pk.h[0] = (_Float16)v.x; pk.h[1] = (_Float16)v.y;
    pk.h[2] = (_Float16)v.z; pk.h[3] = (_Float16)v.w;
    *(uint2*)dstp = pk.u;   // 8B aligned store
}

template<int AMODE, bool GUARD>
__device__ __forceinline__ void load_A_tile(
    _Float16* __restrict__ As, const float* __restrict__ A,
    int m0, int k0, int M, int K, int t,
    const int* __restrict__ rowidx, const int* __restrict__ srcrow,
    const int* __restrict__ maskedf, const float* __restrict__ token)
{
#pragma unroll
    for (int i = 0; i < 4; ++i) {
        int idx4 = t + 256 * i;          // 1024 float4 total
        int r = idx4 >> 3;               // 8 float4 per 32-wide row
        int c = (idx4 & 7) * 4;
        int gr = m0 + r, gc = k0 + c;
        float4 v = make_float4(0.f, 0.f, 0.f, 0.f);
        if (!GUARD || gr < M) {
            if (AMODE == AM_PLAIN) {
                v = *(const float4*)(A + (size_t)gr * K + gc);
            } else if (AMODE == AM_GATHER) {
                v = *(const float4*)(A + (size_t)rowidx[gr] * K + gc);
            } else if (AMODE == AM_POSTOK) {
                v = *(const float4*)(A + (size_t)gr * K + gc);
                if (maskedf[gr]) {
                    float4 tk = *(const float4*)(token + gc);
                    v.x += tk.x; v.y += tk.y; v.z += tk.z; v.w += tk.w;
                }
            } else { // AM_NEGSHF
                int s = srcrow[gr];
                if (s >= 0) v = *(const float4*)(A + (size_t)s * K + gc);
                if (maskedf[gr]) {
                    float4 tk = *(const float4*)(token + gc);
                    v.x += tk.x; v.y += tk.y; v.z += tk.z; v.w += tk.w;
                }
            }
        }
        pk_store(&As[r * LDA_S + c], v);
    }
}

template<int AMODE, int BTRANS, int REDMODE>
__global__ __launch_bounds__(256) void k_gemm(
    const float* __restrict__ A, const float* __restrict__ B,
    float* __restrict__ C,
    int M, int K, int Nc,
    const int* __restrict__ rowidx,
    const int* __restrict__ srcrow, const int* __restrict__ maskedf,
    const float* __restrict__ token,
    const float* __restrict__ bias, const float* __restrict__ prelu_a,
    float* __restrict__ red_acc)
{
    __shared__ _Float16 As[TBM * LDA_S];   // [row][k]  128x32
    __shared__ _Float16 BsT[TBN * LDBT];   // [n][k]    64x32 (transposed!)
    __shared__ float redbuf[256];

    const int tiles_n = (Nc + TBN - 1) / TBN;
    const int bm = (int)blockIdx.x / tiles_n;
    const int bn = (int)blockIdx.x % tiles_n;
    const int m0 = bm * TBM, n0 = bn * TBN;
    const int t = threadIdx.x;
    const int lane = t & 31, wave = t >> 5;
    const bool mfull = (m0 + TBM <= M);

    v8f acc[4];
    acc[0] = v8f{}; acc[1] = v8f{}; acc[2] = v8f{}; acc[3] = v8f{};

    for (int k0 = 0; k0 < K; k0 += TBK) {
        // ---- A tile (128x32): float4 global loads, packed f16 LDS stores
        if (mfull)
            load_A_tile<AMODE, false>(As, A, m0, k0, M, K, t,
                                      rowidx, srcrow, maskedf, token);
        else
            load_A_tile<AMODE, true>(As, A, m0, k0, M, K, t,
                                     rowidx, srcrow, maskedf, token);

        // ---- B tile -> BsT[n][k] (transposed staging)
        if (BTRANS) {
            // B stored [Nc][K]: contiguous along k -> vector load + vector store
#pragma unroll
            for (int i = 0; i < 2; ++i) {
                int idx4 = t + 256 * i;      // 512 float4 total
                int ni = idx4 >> 3;          // 0..63
                int c  = (idx4 & 7) * 4;     // k offset
                int gn = n0 + ni, gk = k0 + c;
                float4 v = make_float4(0.f, 0.f, 0.f, 0.f);
                if (gn < Nc) v = *(const float4*)(B + (size_t)gn * K + gk);
                pk_store(&BsT[ni * LDBT + c], v);
            }
        } else {
            // B stored [K][Nc]: contiguous along n -> vector load, scatter store
#pragma unroll
            for (int i = 0; i < 2; ++i) {
                int idx4 = t + 256 * i;      // 512 float4 total
                int kr = idx4 >> 4;          // 0..31
                int n4 = (idx4 & 15) * 4;
                int gk = k0 + kr;
                float4 v;
                if (n0 + n4 + 3 < Nc) {
                    v = *(const float4*)(B + (size_t)gk * Nc + n0 + n4);
                } else {
                    float tmp[4];
#pragma unroll
                    for (int q = 0; q < 4; ++q) {
                        int gn = n0 + n4 + q;
                        tmp[q] = (gn < Nc) ? B[(size_t)gk * Nc + gn] : 0.f;
                    }
                    v = make_float4(tmp[0], tmp[1], tmp[2], tmp[3]);
                }
                BsT[(n4 + 0) * LDBT + kr] = (_Float16)v.x;
                BsT[(n4 + 1) * LDBT + kr] = (_Float16)v.y;
                BsT[(n4 + 2) * LDBT + kr] = (_Float16)v.z;
                BsT[(n4 + 3) * LDBT + kr] = (_Float16)v.w;
            }
        }
        __syncthreads();

        // ---- A fragment (16x32 f16, ISA 7.12.2 layout): 2 x b128 per lane
        const int lrow  = lane & 15;
        const int khalf = (lane >> 4) * 8;
        const _Float16* ap = &As[(wave * 16 + lrow) * LDA_S];
        v8h alo = *(const v8h*)(ap + khalf);        // 16B aligned
        v8h ahi = *(const v8h*)(ap + 16 + khalf);
        v16h af = __builtin_shufflevector(alo, ahi,
            0,1,2,3,4,5,6,7,8,9,10,11,12,13,14,15);

        // ---- 4 N-subtiles: B fragment contiguous in BsT -> 2 x b128 per lane
        const int kb = (lane >> 4) * 8;
#pragma unroll
        for (int s4 = 0; s4 < 4; ++s4) {
            const _Float16* bp = &BsT[(s4 * 16 + lrow) * LDBT + kb];
            v8h blo = *(const v8h*)(bp);            // 16B aligned
            v8h bhi = *(const v8h*)(bp + 16);
            v16h bf = __builtin_shufflevector(blo, bhi,
                0,1,2,3,4,5,6,7,8,9,10,11,12,13,14,15);
            acc[s4] = __builtin_amdgcn_wmma_f32_16x16x32_f16(
                false, af, false, bf, (short)0, acc[s4], false, false);
        }
        __syncthreads();
    }

    // ---- epilogue: bias / prelu / store or fused -log(sigmoid) reduce
    float local = 0.f;
    const int mrow_off = (lane >> 4) * 8;
#pragma unroll
    for (int s4 = 0; s4 < 4; ++s4) {
        int gn = n0 + s4 * 16 + (lane & 15);
#pragma unroll
        for (int r = 0; r < 8; ++r) {
            int gr = m0 + wave * 16 + mrow_off + r;
            if (gr < M && gn < Nc) {
                float cval = acc[s4][r];
                if (bias) cval += bias[gn];
                if (prelu_a) {
                    float al = prelu_a[0];
                    cval = cval >= 0.f ? cval : al * cval;
                }
                if (REDMODE == RED_NONE) {
                    C[(size_t)gr * Nc + gn] = cval;
                } else {
                    float sg = 1.f / (1.f + __expf(-cval));
                    local += (REDMODE == RED_POS) ? -__logf(sg + 1e-15f)
                                                  : -__logf(1.f - sg + 1e-15f);
                }
            }
        }
    }
    if (REDMODE != RED_NONE) {
        redbuf[t] = local;
        __syncthreads();
        for (int off = 128; off > 0; off >>= 1) {
            if (t < off) redbuf[t] += redbuf[t + off];
            __syncthreads();
        }
        if (t == 0) atomicAdd(red_acc, redbuf[0]);
    }
}

// ---------------------------------------------------------------------------
// Graph / elementwise kernels
// ---------------------------------------------------------------------------
__global__ void k_set_mask(int* maskedf, const int* mask, int n) {
    int i = blockIdx.x * blockDim.x + threadIdx.x;
    if (i < n) maskedf[mask[i]] = 1;
}
__global__ void k_set_srcrow(int* srcrow, const int* keep, const int* shuf, int n) {
    int i = blockIdx.x * blockDim.x + threadIdx.x;
    if (i < n) srcrow[keep[i]] = keep[shuf[i]];
}
__global__ void k_edge_count(const int* src, const int* dst,
                             float* deg, float* rowsum, int E) {
    int e = blockIdx.x * blockDim.x + threadIdx.x;
    if (e < E) {
        atomicAdd(&deg[dst[e]], 1.f);
        atomicAdd(&rowsum[src[e]], 1.f);
    }
}
__global__ void k_dinv(const float* deg, float* dinv, int N) {
    int i = blockIdx.x * blockDim.x + threadIdx.x;
    if (i < N) dinv[i] = rsqrtf(deg[i] + 1.f);
}
__global__ void k_coef(const int* src, const int* dst, const float* dinv,
                       float* coef, int E) {
    int e = blockIdx.x * blockDim.x + threadIdx.x;
    if (e < E) coef[e] = dinv[src[e]] * dinv[dst[e]];
}
__global__ void k_self(const float* h, const float* dinv, float* agg,
                       long long total, int D) {
    long long tt = (long long)blockIdx.x * blockDim.x + threadIdx.x;
    if (tt < total) {
        int i = (int)(tt / D);
        float di = dinv[i];
        agg[tt] = h[tt] * di * di;
    }
}
__global__ void k_scatter(const int* src, const int* dst, const float* coef,
                          const float* h, float* agg, long long total, int D) {
    long long tt = (long long)blockIdx.x * blockDim.x + threadIdx.x;
    if (tt < total) {
        int e = (int)(tt / D), d = (int)(tt % D);
        atomicAdd(&agg[(size_t)dst[e] * D + d],
                  h[(size_t)src[e] * D + d] * coef[e]);
    }
}
__global__ void k_bias_prelu(float* x, const float* b, const float* a,
                             long long total, int D) {
    long long tt = (long long)blockIdx.x * blockDim.x + threadIdx.x;
    if (tt < total) {
        float v = x[tt] + b[(int)(tt % D)];
        float al = a[0];
        x[tt] = v >= 0.f ? v : al * v;
    }
}
__global__ void k_vsum(const int* src, const int* dst, const float* rep,
                       float* vsum, long long total) {
    long long tt = (long long)blockIdx.x * blockDim.x + threadIdx.x;
    if (tt < total) {
        int e = (int)(tt >> 7), d = (int)(tt & 127);
        atomicAdd(&vsum[(size_t)src[e] * 128 + d],
                  rep[(size_t)dst[e] * 128 + d]);
    }
}
__global__ void k_summary(const float* vsum, const float* rowsum,
                          const int* mask, float* summ, int n_mask) {
    long long tt = (long long)blockIdx.x * blockDim.x + threadIdx.x;
    if (tt < (long long)n_mask * 128) {
        int i = (int)(tt >> 7), d = (int)(tt & 127);
        int node = mask[i];
        float s = vsum[(size_t)node * 128 + d] / rowsum[node];
        summ[tt] = 1.f / (1.f + __expf(-s));
    }
}
__global__ void k_zero_rows(float* x, const int* mask, int n_mask, int D) {
    long long tt = (long long)blockIdx.x * blockDim.x + threadIdx.x;
    if (tt < (long long)n_mask * D) {
        int i = (int)(tt / D), d = (int)(tt % D);
        x[(size_t)mask[i] * D + d] = 0.f;
    }
}
__global__ __launch_bounds__(128) void k_cos(const float* REP, const float* RXP,
                                             float* acc, int n_mask) {
    __shared__ float sa[128], sb[128], sd[128];
    int i = blockIdx.x, d = threadIdx.x;
    float x = REP[(size_t)i * 128 + d], y = RXP[(size_t)i * 128 + d];
    sa[d] = x * x; sb[d] = y * y; sd[d] = x * y;
    __syncthreads();
    for (int off = 64; off > 0; off >>= 1) {
        if (d < off) { sa[d] += sa[d+off]; sb[d] += sb[d+off]; sd[d] += sd[d+off]; }
        __syncthreads();
    }
    if (d == 0) {
        float nx = fmaxf(sqrtf(sa[0]), 1e-8f);
        float ny = fmaxf(sqrtf(sb[0]), 1e-8f);
        float c  = sd[0] / (nx * ny);
        atomicAdd(acc, -__logf(1.f - c + 1e-15f));
    }
}
__global__ __launch_bounds__(256) void k_feat(const float* feature, const float* rec,
                                              const int* mask, float* acc) {
    __shared__ float sa[256], sb[256], sd[256];
    int i = blockIdx.x, tt = threadIdx.x;
    int node = mask[i];
    float ax = 0.f, bx = 0.f, dx = 0.f;
    for (int j = tt; j < 1024; j += 256) {
        float x = feature[(size_t)node * 1024 + j];
        float y = rec[(size_t)node * 1024 + j];
        ax += x * x; bx += y * y; dx += x * y;
    }
    sa[tt] = ax; sb[tt] = bx; sd[tt] = dx;
    __syncthreads();
    for (int off = 128; off > 0; off >>= 1) {
        if (tt < off) { sa[tt] += sa[tt+off]; sb[tt] += sb[tt+off]; sd[tt] += sd[tt+off]; }
        __syncthreads();
    }
    if (tt == 0) {
        float nx = fmaxf(sqrtf(sa[0]), 1e-12f);
        float ny = fmaxf(sqrtf(sb[0]), 1e-12f);
        float c  = sd[0] / (nx * ny);
        float df = 1.f - c;
        atomicAdd(acc, df * df);
    }
}
__global__ void k_final(const float* acc, int n_mask, float* out) {
    if (blockIdx.x == 0 && threadIdx.x == 0) {
        float nm = (float)n_mask;
        out[0] = acc[3] / nm;                                       // feat_loss
        out[1] = acc[2] / nm + acc[0] / (nm*nm) + acc[1] / (nm*nm); // dgi_loss
    }
}

// ---------------------------------------------------------------------------
// Host side
// ---------------------------------------------------------------------------
template<int AMODE, int BTRANS, int REDMODE>
static void gemm_t(const float* A, const float* B, float* C,
                   int M, int K, int Nc,
                   const int* rowidx, const int* srcrow,
                   const int* maskedf, const float* token,
                   const float* bias, const float* prelu_a,
                   float* red_acc, hipStream_t stream)
{
    int tiles_m = (M + TBM - 1) / TBM;
    int tiles_n = (Nc + TBN - 1) / TBN;
    k_gemm<AMODE, BTRANS, REDMODE><<<dim3(tiles_m * tiles_n), 256, 0, stream>>>(
        A, B, C, M, K, Nc, rowidx, srcrow, maskedf, token,
        bias, prelu_a, red_acc);
}

static void gemm(const float* A, const float* B, float* C,
                 int M, int K, int Nc,
                 int amode, const int* rowidx, const int* srcrow,
                 const int* maskedf, const float* token,
                 int btrans, const float* bias, const float* prelu_a,
                 int redmode, float* red_acc, hipStream_t stream)
{
    if (redmode == RED_POS) {
        gemm_t<AM_PLAIN,1,RED_POS>(A,B,C,M,K,Nc,rowidx,srcrow,maskedf,token,
                                   bias,prelu_a,red_acc,stream);
    } else if (redmode == RED_NEG) {
        gemm_t<AM_PLAIN,1,RED_NEG>(A,B,C,M,K,Nc,rowidx,srcrow,maskedf,token,
                                   bias,prelu_a,red_acc,stream);
    } else if (btrans) {
        gemm_t<AM_PLAIN,1,RED_NONE>(A,B,C,M,K,Nc,rowidx,srcrow,maskedf,token,
                                    bias,prelu_a,red_acc,stream);
    } else if (amode == AM_GATHER) {
        gemm_t<AM_GATHER,0,RED_NONE>(A,B,C,M,K,Nc,rowidx,srcrow,maskedf,token,
                                     bias,prelu_a,red_acc,stream);
    } else if (amode == AM_POSTOK) {
        gemm_t<AM_POSTOK,0,RED_NONE>(A,B,C,M,K,Nc,rowidx,srcrow,maskedf,token,
                                     bias,prelu_a,red_acc,stream);
    } else if (amode == AM_NEGSHF) {
        gemm_t<AM_NEGSHF,0,RED_NONE>(A,B,C,M,K,Nc,rowidx,srcrow,maskedf,token,
                                     bias,prelu_a,red_acc,stream);
    } else {
        gemm_t<AM_PLAIN,0,RED_NONE>(A,B,C,M,K,Nc,rowidx,srcrow,maskedf,token,
                                    bias,prelu_a,red_acc,stream);
    }
}

static inline unsigned blks(long long total, int tpb) {
    return (unsigned)((total + tpb - 1) / tpb);
}

extern "C" void kernel_launch(void* const* d_in, const int* in_sizes, int n_in,
                              void* d_out, int out_size, void* d_ws, size_t ws_size,
                              hipStream_t stream)
{
    const float* feature   = (const float*)d_in[0];
    const float* pos_token = (const float*)d_in[1];
    const float* neg_token = (const float*)d_in[2];
    const float* w1  = (const float*)d_in[3];
    const float* b1  = (const float*)d_in[4];
    const float* a_enc = (const float*)d_in[5];
    const float* w2  = (const float*)d_in[6];
    const float* b2  = (const float*)d_in[7];
    const float* pw1 = (const float*)d_in[8];
    const float* pb1 = (const float*)d_in[9];
    const float* a_proj = (const float*)d_in[10];
    const float* pw2 = (const float*)d_in[11];
    const float* pb2 = (const float*)d_in[12];
    const float* disc_w = (const float*)d_in[13];
    const float* e2d_w  = (const float*)d_in[14];
    const float* dw  = (const float*)d_in[15];
    const float* db  = (const float*)d_in[16];
    const float* a_dec = (const float*)d_in[17];
    const int* edge_index = (const int*)d_in[18];
    const int* mask_nodes = (const int*)d_in[19];
    const int* keep_nodes = (const int*)d_in[20];
    const int* shuffle    = (const int*)d_in[21];

    const int IN_DIM = 1024, HID = 512, LAT = 128, OUT_DIM = 1024;
    const int N = in_sizes[0] / IN_DIM;
    const int E = in_sizes[18] / 2;
    const int n_mask = in_sizes[19];
    const int n_keep = in_sizes[20];
    const int* src = edge_index;
    const int* dst = edge_index + E;

    // ---- workspace carve-out
    char* p = (char*)d_ws;
    auto alloc = [&](size_t bytes) -> void* {
        void* r = (void*)p;
        p += (bytes + 255) & ~(size_t)255;
        return r;
    };
    float* deg    = (float*)alloc((size_t)N * 4);
    float* rowsum = (float*)alloc((size_t)N * 4);
    float* dinv   = (float*)alloc((size_t)N * 4);
    float* coef   = (float*)alloc((size_t)E * 4);
    int*   maskedf = (int*)alloc((size_t)N * 4);
    int*   srcrow  = (int*)alloc((size_t)N * 4);
    float* acc    = (float*)alloc(4 * 4);               // [pos, neg, cos, feat]
    float* h1     = (float*)alloc((size_t)N * HID * 4);
    float* agg1   = (float*)alloc((size_t)N * HID * 4);
    float* h2     = (float*)alloc((size_t)N * LAT * 4);
    float* rep_pos = (float*)alloc((size_t)N * LAT * 4);
    float* rep_neg = (float*)alloc((size_t)N * LAT * 4);
    float* vsum   = (float*)alloc((size_t)N * LAT * 4);
    float* summ   = (float*)alloc((size_t)n_mask * LAT * 4);
    float* S1     = (float*)alloc((size_t)n_mask * LAT * 4);
    float* P1     = (float*)alloc((size_t)n_mask * LAT * 4);
    float* REP    = (float*)alloc((size_t)n_mask * LAT * 4);
    float* RXP    = (float*)alloc((size_t)n_mask * LAT * 4);
    float* rec0   = (float*)alloc((size_t)N * LAT * 4);
    float* hdec   = (float*)alloc((size_t)N * OUT_DIM * 4);
    float* aggdec = (float*)alloc((size_t)N * OUT_DIM * 4);
    (void)ws_size; (void)n_in; (void)out_size;

    // ---- graph preprocessing
    hipMemsetAsync(maskedf, 0, (size_t)N * 4, stream);
    hipMemsetAsync(srcrow, 0xFF, (size_t)N * 4, stream);   // -1
    hipMemsetAsync(deg, 0, (size_t)N * 4, stream);
    hipMemsetAsync(rowsum, 0, (size_t)N * 4, stream);
    hipMemsetAsync(acc, 0, 4 * 4, stream);
    k_set_mask<<<blks(n_mask,256),256,0,stream>>>(maskedf, mask_nodes, n_mask);
    k_set_srcrow<<<blks(n_keep,256),256,0,stream>>>(srcrow, keep_nodes, shuffle, n_keep);
    k_edge_count<<<blks(E,256),256,0,stream>>>(src, dst, deg, rowsum, E);
    k_dinv<<<blks(N,256),256,0,stream>>>(deg, dinv, N);
    k_coef<<<blks(E,256),256,0,stream>>>(src, dst, dinv, coef, E);

    long long nh = (long long)N * HID, nl = (long long)N * LAT;
    long long eh = (long long)E * HID, el = (long long)E * LAT;
    long long no = (long long)N * OUT_DIM, eo = (long long)E * OUT_DIM;

    // ---- encode(use_pos): GCN1 -> prelu -> GCN2 -> prelu
    gemm(feature, w1, h1, N, IN_DIM, HID, AM_POSTOK, nullptr, nullptr, maskedf,
         pos_token, 0, nullptr, nullptr, RED_NONE, nullptr, stream);
    k_self<<<blks(nh,256),256,0,stream>>>(h1, dinv, agg1, nh, HID);
    k_scatter<<<blks(eh,256),256,0,stream>>>(src, dst, coef, h1, agg1, eh, HID);
    k_bias_prelu<<<blks(nh,256),256,0,stream>>>(agg1, b1, a_enc, nh, HID);
    gemm(agg1, w2, h2, N, HID, LAT, AM_PLAIN, nullptr, nullptr, nullptr,
         nullptr, 0, nullptr, nullptr, RED_NONE, nullptr, stream);
    k_self<<<blks(nl,256),256,0,stream>>>(h2, dinv, rep_pos, nl, LAT);
    k_scatter<<<blks(el,256),256,0,stream>>>(src, dst, coef, h2, rep_pos, el, LAT);
    k_bias_prelu<<<blks(nl,256),256,0,stream>>>(rep_pos, b2, a_enc, nl, LAT);

    // ---- encode(cr): shuffled keep rows + neg_token at mask rows
    gemm(feature, w1, h1, N, IN_DIM, HID, AM_NEGSHF, nullptr, srcrow, maskedf,
         neg_token, 0, nullptr, nullptr, RED_NONE, nullptr, stream);
    k_self<<<blks(nh,256),256,0,stream>>>(h1, dinv, agg1, nh, HID);
    k_scatter<<<blks(eh,256),256,0,stream>>>(src, dst, coef, h1, agg1, eh, HID);
    k_bias_prelu<<<blks(nh,256),256,0,stream>>>(agg1, b1, a_enc, nh, HID);
    gemm(agg1, w2, h2, N, HID, LAT, AM_PLAIN, nullptr, nullptr, nullptr,
         nullptr, 0, nullptr, nullptr, RED_NONE, nullptr, stream);
    k_self<<<blks(nl,256),256,0,stream>>>(h2, dinv, rep_neg, nl, LAT);
    k_scatter<<<blks(el,256),256,0,stream>>>(src, dst, coef, h2, rep_neg, el, LAT);
    k_bias_prelu<<<blks(nl,256),256,0,stream>>>(rep_neg, b2, a_enc, nl, LAT);

    // ---- summary = sigmoid(segment_mean_by_src(rep_pos[dst]))[mask]
    hipMemsetAsync(vsum, 0, (size_t)N * LAT * 4, stream);
    k_vsum<<<blks(el,256),256,0,stream>>>(src, dst, rep_pos, vsum, el);
    k_summary<<<blks((long long)n_mask*LAT,256),256,0,stream>>>(
        vsum, rowsum, mask_nodes, summ, n_mask);

    // ---- projector on mask rows (gather fused into GEMM A-load)
    gemm(rep_pos, pw1, P1, n_mask, LAT, LAT, AM_GATHER, mask_nodes, nullptr,
         nullptr, nullptr, 0, pb1, a_proj, RED_NONE, nullptr, stream);
    gemm(P1, pw2, REP, n_mask, LAT, LAT, AM_PLAIN, nullptr, nullptr, nullptr,
         nullptr, 0, pb2, nullptr, RED_NONE, nullptr, stream);
    gemm(rep_neg, pw1, P1, n_mask, LAT, LAT, AM_GATHER, mask_nodes, nullptr,
         nullptr, nullptr, 0, pb1, a_proj, RED_NONE, nullptr, stream);
    gemm(P1, pw2, RXP, n_mask, LAT, LAT, AM_PLAIN, nullptr, nullptr, nullptr,
         nullptr, 0, pb2, nullptr, RED_NONE, nullptr, stream);

    // ---- discriminator: S1 = summ @ disc_w^T; fused score reductions
    gemm(summ, disc_w, S1, n_mask, LAT, LAT, AM_PLAIN, nullptr, nullptr,
         nullptr, nullptr, /*btrans=*/1, nullptr, nullptr, RED_NONE, nullptr, stream);
    gemm(REP, S1, nullptr, n_mask, LAT, n_mask, AM_PLAIN, nullptr, nullptr,
         nullptr, nullptr, /*btrans=*/1, nullptr, nullptr, RED_POS, acc + 0, stream);
    gemm(RXP, S1, nullptr, n_mask, LAT, n_mask, AM_PLAIN, nullptr, nullptr,
         nullptr, nullptr, /*btrans=*/1, nullptr, nullptr, RED_NEG, acc + 1, stream);

    // ---- cosine loss between projected views
    k_cos<<<n_mask,128,0,stream>>>(REP, RXP, acc + 2, n_mask);

    // ---- decoder: rec = prelu(GCN(zero_mask(rep_pos @ e2d_w), dw, db))
    gemm(rep_pos, e2d_w, rec0, N, LAT, LAT, AM_PLAIN, nullptr, nullptr,
         nullptr, nullptr, 0, nullptr, nullptr, RED_NONE, nullptr, stream);
    k_zero_rows<<<blks((long long)n_mask*LAT,256),256,0,stream>>>(
        rec0, mask_nodes, n_mask, LAT);
    gemm(rec0, dw, hdec, N, LAT, OUT_DIM, AM_PLAIN, nullptr, nullptr,
         nullptr, nullptr, 0, nullptr, nullptr, RED_NONE, nullptr, stream);
    k_self<<<blks(no,256),256,0,stream>>>(hdec, dinv, aggdec, no, OUT_DIM);
    k_scatter<<<blks(eo,256),256,0,stream>>>(src, dst, coef, hdec, aggdec, eo, OUT_DIM);
    k_bias_prelu<<<blks(no,256),256,0,stream>>>(aggdec, db, a_dec, no, OUT_DIM);

    // ---- feature cosine reconstruction loss on mask rows
    k_feat<<<n_mask,256,0,stream>>>(feature, aggdec, mask_nodes, acc + 3);

    // ---- combine
    k_final<<<1,1,0,stream>>>(acc, n_mask, (float*)d_out);
}
